// TorusGraphs_40243843564077
// MI455X (gfx1250) — matrix operand order repaired and checked by reading, not compiled, not run on hardware
//
#include <hip/hip_runtime.h>

typedef _Float16 v16h __attribute__((ext_vector_type(16)));
typedef _Float16 v8h  __attribute__((ext_vector_type(8)));
typedef float    v8f  __attribute__((ext_vector_type(8)));

#define PP   128
#define ZZ   8128
#define NS   20000

// ---------------------------------------------------------------------------
// Kernel 0: zero the scalar output (harness poisons d_out before timing).
// ---------------------------------------------------------------------------
__global__ void tg_zero_out(float* out) { out[0] = 0.0f; }

// ---------------------------------------------------------------------------
// M[r][c], r,c in [0,256):  M = [[A/2, B^T/2],[B/2, A/2]]
//   A[i][j] = theta_c[z(i,j)] (sym, zero diag)
//   B[i][j] = +theta_s[z]  (i<j),  -theta_s[z] (i>j), zero diag
// so that lp[n] = u_n^T M u_n with u = [cos(x); sin(x)].
// ---------------------------------------------------------------------------
__device__ __forceinline__ float tg_m_entry(int r, int c,
                                            const float* __restrict__ thc,
                                            const float* __restrict__ ths) {
    int rb = r & (PP - 1), cb = c & (PP - 1);
    if (rb == cb) return 0.0f;
    int rH = r >> 7, cH = c >> 7;            // r,c in [0,256) -> 0 or 1
    int i  = rb < cb ? rb : cb;
    int j  = rb < cb ? cb : rb;
    int z  = i * PP - (i * (i + 1)) / 2 + (j - i - 1);   // triu_indices(P,1) order
    if (rH == cH) return 0.5f * thc[z];                  // A/2 blocks
    int sinIdx   = rH ? rb : cb;                         // index on the sin side
    int otherIdx = rH ? cb : rb;
    float sgn = (sinIdx < otherIdx) ? 0.5f : -0.5f;
    return sgn * ths[z];
}

// ---------------------------------------------------------------------------
// Kernel 1: build M (256x256 f16) pre-packed in WMMA B-fragment order.
// B 32x16 f16 layout used here (contiguous k, same as A): tile(k,j), lane l, vgpr v holds halves
//   K0 = 32k + 16*(l>=16) + 2v , K1 = K0+1 , N = 16j + (l%16)
// K = 256 -> 8 k-tiles; N = 256 -> 16 j-tiles; 128 tiles * 1KB = 128 KB.
// One thread produces one packed dword (32768 dwords).
// ---------------------------------------------------------------------------
__global__ void tg_build_bpk(const float* __restrict__ theta,
                             _Float16* __restrict__ bpk) {
    int idx  = blockIdx.x * blockDim.x + threadIdx.x;    // 0 .. 32767
    int v    = idx & 7;
    int lane = (idx >> 3) & 31;
    int tile = idx >> 8;                                 // 0 .. 127
    int j    = tile & 15;
    int k    = tile >> 4;                                // 0 .. 7
    int hi   = lane >> 4;
    int ncol = 16 * j + (lane & 15);
    int k0   = 32 * k + 16 * hi + 2 * v;                 // in [0, 256)
    const float* thc = theta;        // theta[0,0,:]
    const float* ths = theta + ZZ;   // theta[0,1,:]
    bpk[2 * idx + 0] = (_Float16)tg_m_entry(k0,     ncol, thc, ths);
    bpk[2 * idx + 1] = (_Float16)tg_m_entry(k0 + 1, ncol, thc, ths);
}

// ---------------------------------------------------------------------------
// Kernel 2: fused trig + WMMA GEMM + row-dot + logaddexp reduction.
// blockDim = 64 (2 wave32s); each wave owns TWO 16-row sample blocks
// (32 samples), so every B fragment loaded feeds two v_wmma ops.
// grid = (ceil(625/2)=313, 2): y=0 -> data pass (X, log_nx),
//                              y=1 -> noise pass (noise, log_ny).
// ---------------------------------------------------------------------------
__global__ __launch_bounds__(64)
void tg_nce_pass(const float* __restrict__ Xd,  const float* __restrict__ Nd,
                 const float* __restrict__ log_nx, const float* __restrict__ log_ny,
                 const float* __restrict__ logc, const v16h* __restrict__ bpk,
                 float* __restrict__ out) {
    __shared__ _Float16 Ush[2][32][256];   // per-wave u = [cos | sin], f16 (32 KB)
    __shared__ float    blockSum;

    const int  mode = blockIdx.y;
    const float* ang = mode ? Nd     : Xd;
    const float* aux = mode ? log_ny : log_nx;

    const int lane = threadIdx.x & 31;
    const int w    = threadIdx.x >> 5;
    const int sampleBase = (blockIdx.x * 2 + w) * 32;

    if (threadIdx.x == 0) blockSum = 0.0f;

    // ---- stage 1: sincos into LDS (32 rows x 128 angles per wave) ----
    for (int e = lane; e < 32 * PP; e += 32) {
        int row = e >> 7, col = e & (PP - 1);
        int sample = sampleBase + row;
        float s = 0.0f, c = 0.0f;
        if (sample < NS) {
            float a = ang[sample * PP + col];
            __sincosf(a, &s, &c);
        }
        Ush[w][row][col]      = (_Float16)c;
        Ush[w][row][PP + col] = (_Float16)s;
    }
    __syncthreads();

    // ---- stage 2: load all A fragments (two 16-row blocks, 8 k-tiles) ----
    // A 16x32 f16 layout: lane's M = lane%16; 16 contiguous K at 32k+16*hi.
    const int hi    = lane >> 4;
    const int arow  = lane & 15;
    const int hiOff = hi * 16;
    v16h a0[8], a1[8];
    #pragma unroll
    for (int k = 0; k < 8; ++k) {
        const _Float16* r0 = &Ush[w][arow][32 * k + hiOff];
        const _Float16* r1 = &Ush[w][16 + arow][32 * k + hiOff];
        v8h l0 = *(const v8h*)r0;
        v8h h0 = *(const v8h*)(r0 + 8);
        v8h l1 = *(const v8h*)r1;
        v8h h1 = *(const v8h*)(r1 + 8);
        a0[k] = __builtin_shufflevector(l0, h0, 0,1,2,3,4,5,6,7,8,9,10,11,12,13,14,15);
        a1[k] = __builtin_shufflevector(l1, h1, 0,1,2,3,4,5,6,7,8,9,10,11,12,13,14,15);
    }

    // ---- stage 3: T = U * M via v_wmma_f32_16x16x32_f16, fused diag dot ----
    float lp0[8] = {0.f,0.f,0.f,0.f,0.f,0.f,0.f,0.f};
    float lp1[8] = {0.f,0.f,0.f,0.f,0.f,0.f,0.f,0.f};

    for (int j = 0; j < 16; ++j) {            // output column tile
        v8f acc0 = {0.f,0.f,0.f,0.f,0.f,0.f,0.f,0.f};
        v8f acc1 = {0.f,0.f,0.f,0.f,0.f,0.f,0.f,0.f};
        #pragma unroll
        for (int k = 0; k < 8; ++k) {         // K = 256 -> 8 tiles of 32
            v16h b = bpk[(k * 16 + j) * 32 + lane];   // pre-packed fragment
            acc0 = __builtin_amdgcn_wmma_f32_16x16x32_f16(
                       false, a0[k], false, b, (short)0, acc0, false, false);
            acc1 = __builtin_amdgcn_wmma_f32_16x16x32_f16(
                       false, a1[k], false, b, (short)0, acc1, false, false);
        }
        // C/D layout: element (m = v + 8*hi, n = lane%16) -> lp[m] += T*U[m, 16j+n]
        int col = 16 * j + (lane & 15);
        #pragma unroll
        for (int v = 0; v < 8; ++v) {
            lp0[v] += acc0[v] * (float)Ush[w][v + 8 * hi][col];
            lp1[v] += acc1[v] * (float)Ush[w][16 + v + 8 * hi][col];
        }
    }

    // ---- stage 4: reduce over the 16 lanes of each half (wave32 butterfly) ----
    #pragma unroll
    for (int v = 0; v < 8; ++v) {
        float t0 = lp0[v], t1 = lp1[v];
        t0 += __shfl_xor(t0, 1, 32);  t1 += __shfl_xor(t1, 1, 32);
        t0 += __shfl_xor(t0, 2, 32);  t1 += __shfl_xor(t1, 2, 32);
        t0 += __shfl_xor(t0, 4, 32);  t1 += __shfl_xor(t1, 4, 32);
        t0 += __shfl_xor(t0, 8, 32);  t1 += __shfl_xor(t1, 8, 32);
        lp0[v] = t0;  lp1[v] = t1;
    }

    // ---- stage 5: per-sample objective term, block accumulate ----
    if ((lane & 15) == 0) {                   // lanes 0 (rows v+0), 16 (rows v+8)
        const float logN  = __logf((float)NS);   // n == m == 20000
        const float lc    = logc[0];
        const float scale = 1.0f / (float)NS;
        float local = 0.0f;
        #pragma unroll
        for (int v = 0; v < 8; ++v) {
            #pragma unroll
            for (int blk = 0; blk < 2; ++blk) {
                int sample = sampleBase + 16 * blk + 8 * hi + v;
                if (sample < NS) {
                    float lp = blk ? lp1[v] : lp0[v];
                    float a  = logN + lp;             // model side
                    float b  = logN + aux[sample];    // noise density side
                    float mx = fmaxf(a, b), mn = fminf(a, b);
                    float lae = mx + log1pf(__expf(mn - mx));
                    float numer = logN + lc + (mode ? aux[sample] : lp);
                    local += numer - lae;
                }
            }
        }
        atomicAdd(&blockSum, local * scale);      // ds_add_f32
    }
    __syncthreads();
    if (threadIdx.x == 0) atomicAdd(out, blockSum);
}

// ---------------------------------------------------------------------------
// Launch
// ---------------------------------------------------------------------------
extern "C" void kernel_launch(void* const* d_in, const int* in_sizes, int n_in,
                              void* d_out, int out_size, void* d_ws, size_t ws_size,
                              hipStream_t stream) {
    const float* X      = (const float*)d_in[0];   // [N, 128]
    const float* noise  = (const float*)d_in[1];   // [N, 128]
    const float* log_nx = (const float*)d_in[2];   // [N]
    const float* log_ny = (const float*)d_in[3];   // [N]
    const float* theta  = (const float*)d_in[4];   // [1, 2, Z] flat
    const float* logc   = (const float*)d_in[5];   // [1]
    float* out = (float*)d_out;

    _Float16* bpk = (_Float16*)d_ws;               // 128 KB packed B fragments

    tg_zero_out<<<1, 1, 0, stream>>>(out);
    tg_build_bpk<<<128, 256, 0, stream>>>(theta, bpk);   // 32768 dwords

    dim3 grid(313, 2);                             // 313*2 waves >= 625 blocks of 32
    tg_nce_pass<<<grid, 64, 0, stream>>>(X, noise, log_nx, log_ny, logc,
                                         (const v16h*)d_ws, out);
}
